// MixtralSparseMoELayer_30262339567730
// MI455X (gfx1250) — matrix operand order repaired and checked by reading, not compiled, not run on hardware
//
#include <hip/hip_runtime.h>
#include <hip/hip_bf16.h>

#define T_TOK 4096
#define H_DIM 1024
#define F_DIM 3584
#define E_NUM 8
#define K_TOP 2

typedef __attribute__((ext_vector_type(16))) __bf16 v16bf;
typedef __attribute__((ext_vector_type(8)))  float  v8f;

__device__ __forceinline__ __bf16 tobf(float f) { return (__bf16)f; }
__device__ __forceinline__ v8f v8f_zero() {
  v8f z;
#pragma unroll
  for (int i = 0; i < 8; ++i) z[i] = 0.f;
  return z;
}

// fast silu: g * rcp(1+exp(-g)) -> v_exp_f32 + v_rcp_f32 (TRANS pipe,
// co-executes with WMMA) instead of the IEEE v_div_scale/fixup sequence.
__device__ __forceinline__ float fast_silu(float g) {
  return g * __builtin_amdgcn_rcpf(1.f + __expf(-g));
}

// ---------------- zero output accumulation region + expert counters ----------
__global__ void zero_kernel(float* __restrict__ out, int n4, int* __restrict__ cnt) {
  int i = blockIdx.x * blockDim.x + threadIdx.x;
  float4 z = make_float4(0.f, 0.f, 0.f, 0.f);
  for (int j = i; j < n4; j += gridDim.x * blockDim.x)
    ((float4*)out)[j] = z;
  if (i < E_NUM) cnt[i] = 0;
}

// ---------------- router: one wave32 per token ------------------------------
__global__ void router_kernel(const float* __restrict__ x,
                              const float* __restrict__ gw,
                              const float* __restrict__ pbias,
                              const float* __restrict__ temp,
                              float* __restrict__ d_out,
                              int* __restrict__ cnt,
                              int* __restrict__ list,
                              float* __restrict__ wlist) {
  const int wave = threadIdx.x >> 5;
  const int lane = threadIdx.x & 31;
  const int t = blockIdx.x * (blockDim.x >> 5) + wave;
  if (t >= T_TOK) return;

  float p[E_NUM] = {0.f};
  const float* xr = x + (size_t)t * H_DIM;
  for (int h = lane; h < H_DIM; h += 32) {
    float xv = xr[h];
    const float* g = gw + h * E_NUM;
#pragma unroll
    for (int e = 0; e < E_NUM; ++e) p[e] += xv * g[e];
  }
#pragma unroll
  for (int e = 0; e < E_NUM; ++e)
    for (int off = 16; off > 0; off >>= 1)
      p[e] += __shfl_xor(p[e], off, 32);

  if (lane == 0) {
    float logit[E_NUM];
#pragma unroll
    for (int e = 0; e < E_NUM; ++e) {
      float tm = fminf(fmaxf(temp[e], 0.1f), 10.f);
      logit[e] = (p[e] + pbias[e]) / tm;
    }
    int i0 = 0;
#pragma unroll
    for (int e = 1; e < E_NUM; ++e) if (logit[e] > logit[i0]) i0 = e;
    int i1 = (i0 == 0) ? 1 : 0;
#pragma unroll
    for (int e = 0; e < E_NUM; ++e)
      if (e != i0 && logit[e] > logit[i1]) i1 = e;

    float ex = __expf(logit[i1] - logit[i0]);   // <= 1
    float w0 = 1.f / (1.f + ex);
    float w1 = ex * w0;

    float* rw = d_out + (size_t)T_TOK * H_DIM;
    int*  sel = (int*)(rw + (size_t)T_TOK * K_TOP);
    rw[t * K_TOP + 0] = w0;
    rw[t * K_TOP + 1] = w1;
    sel[t * K_TOP + 0] = i0;
    sel[t * K_TOP + 1] = i1;

    int s0 = atomicAdd(&cnt[i0], 1);
    list[i0 * T_TOK + s0] = t; wlist[i0 * T_TOK + s0] = w0;
    int s1 = atomicAdd(&cnt[i1], 1);
    list[i1 * T_TOK + s1] = t; wlist[i1 * T_TOK + s1] = w1;
  }
}

// ---------------- routed expert FFN: 16-token tile per block ----------------
__launch_bounds__(256)
__global__ void expert_ffn_kernel(const float* __restrict__ x,
                                  const float* __restrict__ w1,
                                  const float* __restrict__ v,
                                  const float* __restrict__ w2,
                                  const int* __restrict__ cnt,
                                  const int* __restrict__ list,
                                  const float* __restrict__ wlist,
                                  float* __restrict__ out) {
  __shared__ __bf16 xbf[16 * H_DIM];   // token tile, bf16, 32 KB
  __shared__ __bf16 actb[16 * 128];    // silu(g)*u for one F-block, 4 KB
  __shared__ int   tok[16];
  __shared__ float tw[16];

  const int e = blockIdx.x >> 8;            // 256 tiles per expert
  const int tile = blockIdx.x & 255;
  const int count = cnt[e];
  const int tileStart = tile * 16;
  if (tileStart >= count) return;           // block-uniform exit
  const int nTok = min(16, count - tileStart);

  const int tid = threadIdx.x;
  const int wave = tid >> 5;                // 8 waves
  const int lane = tid & 31;
  const int nl = lane & 15;                 // column within 16 (B/C/D), row M (A)
  const int khalf = lane >> 4;              // K-half select

  if (tid < 16) {
    tok[tid] = (tid < nTok) ? list[e * T_TOK + tileStart + tid] : 0;
    tw[tid]  = (tid < nTok) ? wlist[e * T_TOK + tileStart + tid] : 0.f;
  }
  __syncthreads();

  for (int i = tid; i < 16 * H_DIM; i += 256) {
    int m = i >> 10;
    int h = i & (H_DIM - 1);
    float xv = (m < nTok) ? x[(size_t)tok[m] * H_DIM + h] : 0.f;
    xbf[i] = tobf(xv);
  }
  __syncthreads();

  const float* w1e = w1 + (size_t)e * H_DIM * F_DIM;
  const float* ve  = v  + (size_t)e * H_DIM * F_DIM;
  const float* w2e = w2 + (size_t)e * F_DIM * H_DIM;

  v8f acc[8];
#pragma unroll
  for (int i = 0; i < 8; ++i) acc[i] = v8f_zero();

  for (int fb = 0; fb < F_DIM; fb += 128) {
    // -------- Phase A: act[:, fb + wave*16 .. +15] = silu(X@w1) * (X@v) -----
    const int fcol = fb + wave * 16 + nl;
    v8f g = v8f_zero(), u = v8f_zero();
    for (int kb = 0; kb < H_DIM; kb += 32) {
      v16bf a, bg, bu;
#pragma unroll
      for (int j = 0; j < 8; ++j) {          // A: 16x32 bf16, contiguous k = 16*khalf + s
        int k = kb + khalf * 16 + j * 2;
        a[2 * j]     = xbf[(nl << 10) + k];
        a[2 * j + 1] = xbf[(nl << 10) + k + 1];
      }
      const int krow = kb + khalf * 16;      // B: 32x16, K-half by lane>>4
#pragma unroll
      for (int j = 0; j < 8; ++j) {
        const float* p1 = w1e + (size_t)(krow + 2 * j) * F_DIM + fcol;
        bg[2 * j]     = tobf(p1[0]);
        bg[2 * j + 1] = tobf(p1[F_DIM]);
        const float* p2 = ve + (size_t)(krow + 2 * j) * F_DIM + fcol;
        bu[2 * j]     = tobf(p2[0]);
        bu[2 * j + 1] = tobf(p2[F_DIM]);
      }
      __builtin_prefetch(w1e + (size_t)(krow + 32) * F_DIM + fcol, 0, 1);
      g = __builtin_amdgcn_wmma_f32_16x16x32_bf16(false, a, false, bg, (short)0, g, false, false);
      u = __builtin_amdgcn_wmma_f32_16x16x32_bf16(false, a, false, bu, (short)0, u, false, false);
    }
#pragma unroll
    for (int r = 0; r < 8; ++r) {            // C layout: row = r + 8*khalf
      int m = khalf * 8 + r;
      float hv = fast_silu(g[r]) * u[r];
      actb[m * 128 + wave * 16 + nl] = tobf(hv);
    }
    __syncthreads();

    // -------- Phase B: out[:, wave*128 .. +127] += act @ w2[fb..fb+127, :] --
    for (int kk = 0; kk < 128; kk += 32) {
      v16bf a;
#pragma unroll
      for (int j = 0; j < 8; ++j) {
        int k = kk + khalf * 16 + j * 2;
        a[2 * j]     = actb[nl * 128 + k];
        a[2 * j + 1] = actb[nl * 128 + k + 1];
      }
      const int krow = fb + kk + khalf * 16;
#pragma unroll
      for (int t8 = 0; t8 < 8; ++t8) {
        const int ncol = wave * 128 + t8 * 16 + nl;
        v16bf b;
#pragma unroll
        for (int j = 0; j < 8; ++j) {
          const float* p = w2e + (size_t)(krow + 2 * j) * H_DIM + ncol;
          b[2 * j]     = tobf(p[0]);
          b[2 * j + 1] = tobf(p[H_DIM]);
        }
        acc[t8] = __builtin_amdgcn_wmma_f32_16x16x32_bf16(false, a, false, b, (short)0, acc[t8], false, false);
      }
    }
    __syncthreads();
  }

  // -------- Epilogue: scale by routing weight, accumulate into output -------
#pragma unroll
  for (int t8 = 0; t8 < 8; ++t8) {
    const int ncol = wave * 128 + t8 * 16 + nl;
#pragma unroll
    for (int r = 0; r < 8; ++r) {
      int m = khalf * 8 + r;
      if (m < nTok)
        atomicAdd(&out[(size_t)tok[m] * H_DIM + ncol], acc[t8][r] * tw[m]);
    }
  }
}

// ---------------------------------------------------------------------------
extern "C" void kernel_launch(void* const* d_in, const int* in_sizes, int n_in,
                              void* d_out, int out_size, void* d_ws, size_t ws_size,
                              hipStream_t stream) {
  const float* x     = (const float*)d_in[0];  // [B,S,H]
  const float* gw    = (const float*)d_in[1];  // [H,E]
  const float* w1    = (const float*)d_in[2];  // [E,H,F]
  const float* v     = (const float*)d_in[3];  // [E,H,F]
  const float* w2    = (const float*)d_in[4];  // [E,F,H]
  const float* pbias = (const float*)d_in[5];  // [E]
  const float* temp  = (const float*)d_in[6];  // [E]
  float* out = (float*)d_out;                  // output | routing_weights | selected

  int*   cnt   = (int*)d_ws;                         // 8 counters (64-int pad)
  int*   list  = cnt + 64;                           // E*T token ids
  float* wlist = (float*)(list + E_NUM * T_TOK);     // E*T routing weights

  zero_kernel<<<1024, 256, 0, stream>>>(out, (T_TOK * H_DIM) / 4, cnt);
  router_kernel<<<T_TOK / 8, 256, 0, stream>>>(x, gw, pbias, temp, out, cnt, list, wlist);
  expert_ffn_kernel<<<E_NUM * (T_TOK / 16), 256, 0, stream>>>(x, w1, v, w2, cnt, list, wlist, out);
}
